// MemoryEfficientAttention_60919816126702
// MI455X (gfx1250) — matrix operand order repaired and checked by reading, not compiled, not run on hardware
//
#include <hip/hip_runtime.h>

// ---------------------------------------------------------------------------
// FlashAttention-2 forward for MI455X (gfx1250, wave32, WMMA f16 16x16x32).
// B=2, H=16, S=2048, D=128, fp32 in/out. f16 WMMA w/ fp32 accumulation.
// Round 3: double-buffered LDS staging (1 barrier/step), grouped B-fragment
// loads (4 in flight per WMMA group), packed V-transpose staging stores,
// DPP ROW_XMASK softmax reductions.
// ---------------------------------------------------------------------------

typedef __attribute__((ext_vector_type(16))) _Float16 v16h;
typedef __attribute__((ext_vector_type(4)))  _Float16 v4h;
typedef __attribute__((ext_vector_type(2)))  _Float16 v2h;
typedef __attribute__((ext_vector_type(8)))  float    v8f;

#define FA_B      2
#define FA_H      16
#define FA_S      2048
#define FA_D      128
#define WAVES     8
#define NTHREADS  (WAVES * 32)     // 256
#define QTILE     (WAVES * 16)     // 128 query rows per workgroup
#define KSTEP     64               // keys per main-loop step (4 x 16 sub-tiles)
#define NSTEPS    (FA_S / KSTEP)   // 32

// 1/sqrt(128) * log2(e), folded into Q at f16-conversion time
#define CSCALE    (0.08838834764831845f * 1.4426950408889634f)

// ---- 16-lane butterflies via DPP ROW_XMASK (pure VALU, no LDS traffic) ----
__device__ __forceinline__ float rowmax16(float v) {
  v = fmaxf(v, __int_as_float(__builtin_amdgcn_update_dpp(
          0, __float_as_int(v), 0x161, 0xf, 0xf, true)));   // xor 1
  v = fmaxf(v, __int_as_float(__builtin_amdgcn_update_dpp(
          0, __float_as_int(v), 0x162, 0xf, 0xf, true)));   // xor 2
  v = fmaxf(v, __int_as_float(__builtin_amdgcn_update_dpp(
          0, __float_as_int(v), 0x164, 0xf, 0xf, true)));   // xor 4
  v = fmaxf(v, __int_as_float(__builtin_amdgcn_update_dpp(
          0, __float_as_int(v), 0x168, 0xf, 0xf, true)));   // xor 8
  return v;
}
__device__ __forceinline__ float rowsum16(float v) {
  v += __int_as_float(__builtin_amdgcn_update_dpp(
          0, __float_as_int(v), 0x161, 0xf, 0xf, true));
  v += __int_as_float(__builtin_amdgcn_update_dpp(
          0, __float_as_int(v), 0x162, 0xf, 0xf, true));
  v += __int_as_float(__builtin_amdgcn_update_dpp(
          0, __float_as_int(v), 0x164, 0xf, 0xf, true));
  v += __int_as_float(__builtin_amdgcn_update_dpp(
          0, __float_as_int(v), 0x168, 0xf, 0xf, true));
  return v;
}

// 32B per-lane fragment slice (compiler merges to 2x ds_load_b128).
__device__ __forceinline__ v16h load_half16(const _Float16* p) {
  v16h r;
  #pragma unroll
  for (int i = 0; i < 8; ++i) {
    v2h q = *(const v2h*)(p + 2 * i);
    r[2*i] = q.x; r[2*i + 1] = q.y;
  }
  return r;
}

// Stage one 64x128 fp32 K/V tile into f16 LDS (K row-major, V transposed,
// V stores packed as key-pairs -> b32 DS stores).
__device__ __forceinline__ void stage_tile(const float* __restrict__ ks,
                                           const float* __restrict__ vs,
                                           _Float16* Ksb, _Float16* Vtb,
                                           int tid) {
  #pragma unroll
  for (int e = tid * 4; e < KSTEP * FA_D; e += NTHREADS * 4) {
    float4 f = *(const float4*)(ks + e);
    v4h hk = { (_Float16)f.x, (_Float16)f.y, (_Float16)f.z, (_Float16)f.w };
    *(v4h*)(Ksb + e) = hk;
  }
  #pragma unroll
  for (int e = tid * 4; e < (KSTEP / 2) * FA_D; e += NTHREADS * 4) {
    const int kp  = e >> 7;        // key-pair index (0..31)
    const int dim = e & 127;       // dim base (multiple of 4)
    const float* r0 = vs + (size_t)(2 * kp) * FA_D + dim;
    float4 a = *(const float4*)(r0);
    float4 b = *(const float4*)(r0 + FA_D);
    v2h p0 = { (_Float16)a.x, (_Float16)b.x };
    v2h p1 = { (_Float16)a.y, (_Float16)b.y };
    v2h p2 = { (_Float16)a.z, (_Float16)b.z };
    v2h p3 = { (_Float16)a.w, (_Float16)b.w };
    *(v2h*)(Vtb + (dim + 0) * KSTEP + 2 * kp) = p0;
    *(v2h*)(Vtb + (dim + 1) * KSTEP + 2 * kp) = p1;
    *(v2h*)(Vtb + (dim + 2) * KSTEP + 2 * kp) = p2;
    *(v2h*)(Vtb + (dim + 3) * KSTEP + 2 * kp) = p3;
  }
}

__global__ __launch_bounds__(NTHREADS, 1)
void fa2_fwd_gfx1250(const float* __restrict__ Q, const float* __restrict__ K,
                     const float* __restrict__ V, float* __restrict__ O)
{
  __shared__ _Float16 Ks[2][KSTEP * FA_D];         // 2 x 16 KB
  __shared__ _Float16 Vt[2][FA_D * KSTEP];         // 2 x 16 KB
  __shared__ _Float16 Ps[WAVES][16 * KSTEP];       // 16 KB  (80 KB total)

  const int tid  = threadIdx.x;
  const int wave = tid >> 5;
  const int lane = tid & 31;
  const int bh   = blockIdx.y;
  const int qblk = blockIdx.x;

  const size_t slab = (size_t)bh * FA_S * FA_D;
  const float* Qb = Q + slab;
  const float* Kb = K + slab;
  const float* Vb = V + slab;
  float*       Ob = O + slab;

  const int q0  = qblk * QTILE + wave * 16;
  const int mA  = lane & 15;            // A-fragment row (M)
  const int kbA = lane & 16;            // A: K-base 0 / 16 per lane half
  const int nB  = lane & 15;            // B-fragment column (N)
  const int kbB = lane & 16;            // B: K-base 0 / 16 per lane half
  const int hi  = lane >> 4;            // C-layout row half (M += 8*hi)

  // ---- Q tile: 4 A-fragments, fp32 -> f16 with softmax scale folded in ----
  v16h qa[4];
  #pragma unroll
  for (int t = 0; t < 4; ++t) {
    const float* row = Qb + (size_t)(q0 + mA) * FA_D + t * 32;
    #pragma unroll
    for (int i = 0; i < 4; ++i) {
      float2 f = *(const float2*)(row + kbA + 2 * i);
      qa[t][2*i]     = (_Float16)(f.x * CSCALE);
      qa[t][2*i + 1] = (_Float16)(f.y * CSCALE);
    }
    #pragma unroll
    for (int i = 0; i < 4; ++i) {
      float2 f = *(const float2*)(row + 8 + kbA + 2 * i);
      qa[t][8 + 2*i]     = (_Float16)(f.x * CSCALE);
      qa[t][8 + 2*i + 1] = (_Float16)(f.y * CSCALE);
    }
  }

  v8f zero = {};
  v8f acc[8];
  float mrow[8], lrow[8];
  #pragma unroll
  for (int i = 0; i < 8; ++i) { acc[i] = zero; mrow[i] = -1e30f; lrow[i] = 0.0f; }

  _Float16* Pw = Ps[wave];

  // ---- Prologue: stage tile 0, prefetch tile 1 ----
  stage_tile(Kb, Vb, Ks[0], Vt[0], tid);
  __builtin_prefetch(Kb + (size_t)KSTEP * FA_D + tid * 32, 0, 1);
  __builtin_prefetch(Vb + (size_t)KSTEP * FA_D + tid * 32, 0, 1);
  __syncthreads();

  // =========================== main key loop ===============================
  for (int s = 0; s < NSTEPS; ++s) {
    const int cur = s & 1;

    // ---- Stage tile s+1 into the other buffer; prefetch tile s+2 ----
    if (s + 1 < NSTEPS) {
      stage_tile(Kb + (size_t)(s + 1) * KSTEP * FA_D,
                 Vb + (size_t)(s + 1) * KSTEP * FA_D,
                 Ks[cur ^ 1], Vt[cur ^ 1], tid);
      if (s + 2 < NSTEPS) {
        __builtin_prefetch(Kb + (size_t)(s + 2) * KSTEP * FA_D + tid * 32, 0, 1);
        __builtin_prefetch(Vb + (size_t)(s + 2) * KSTEP * FA_D + tid * 32, 0, 1);
      }
    }

    const _Float16* kf = &Ks[cur][nB * FA_D + kbB];
    const _Float16* vf = &Vt[cur][nB * KSTEP + kbB];

    // ---- S = Q*K^T : per d-chunk, load 4 B-frags then 4 WMMAs ----
    v8f sc[4];
    #pragma unroll
    for (int h = 0; h < 4; ++h) sc[h] = zero;
    #pragma unroll
    for (int t = 0; t < 4; ++t) {
      v16h b0 = load_half16(kf + 0 * 16 * FA_D + t * 32);
      v16h b1 = load_half16(kf + 1 * 16 * FA_D + t * 32);
      v16h b2 = load_half16(kf + 2 * 16 * FA_D + t * 32);
      v16h b3 = load_half16(kf + 3 * 16 * FA_D + t * 32);
      sc[0] = __builtin_amdgcn_wmma_f32_16x16x32_f16(
                  false, qa[t], false, b0, (short)0, sc[0], false, false);
      sc[1] = __builtin_amdgcn_wmma_f32_16x16x32_f16(
                  false, qa[t], false, b1, (short)0, sc[1], false, false);
      sc[2] = __builtin_amdgcn_wmma_f32_16x16x32_f16(
                  false, qa[t], false, b2, (short)0, sc[2], false, false);
      sc[3] = __builtin_amdgcn_wmma_f32_16x16x32_f16(
                  false, qa[t], false, b3, (short)0, sc[3], false, false);
    }

    // ---- Online softmax: per C-row, DPP butterflies ----
    #pragma unroll
    for (int r = 0; r < 8; ++r) {
      float s0 = sc[0][r], s1 = sc[1][r], s2 = sc[2][r], s3 = sc[3][r];
      float mx   = rowmax16(fmaxf(fmaxf(s0, s1), fmaxf(s2, s3)));
      float mnew = fmaxf(mrow[r], mx);
      float alpha = exp2f(mrow[r] - mnew);
      float e0 = exp2f(s0 - mnew);
      float e1 = exp2f(s1 - mnew);
      float e2 = exp2f(s2 - mnew);
      float e3 = exp2f(s3 - mnew);
      float rs = rowsum16((e0 + e1) + (e2 + e3));
      lrow[r] = fmaf(lrow[r], alpha, rs);
      mrow[r] = mnew;
      #pragma unroll
      for (int nt = 0; nt < 8; ++nt) acc[nt][r] *= alpha;
      _Float16* pr = &Pw[(r + 8 * hi) * KSTEP + nB];
      pr[0]  = (_Float16)e0;
      pr[16] = (_Float16)e1;
      pr[32] = (_Float16)e2;
      pr[48] = (_Float16)e3;
    }
    asm volatile("s_wait_dscnt 0" ::: "memory");  // same-wave LDS RAW fence

    // ---- P as two A-fragments (keys 0-31 / 32-63) ----
    v16h pa[2];
    #pragma unroll
    for (int g = 0; g < 2; ++g) {
      const _Float16* prow = &Pw[mA * KSTEP + g * 32];
      #pragma unroll
      for (int i = 0; i < 4; ++i) {
        v2h p = *(const v2h*)(prow + kbA + 2 * i);
        pa[g][2*i] = p.x; pa[g][2*i + 1] = p.y;
      }
      #pragma unroll
      for (int i = 0; i < 4; ++i) {
        v2h p = *(const v2h*)(prow + 8 + kbA + 2 * i);
        pa[g][8 + 2*i] = p.x; pa[g][8 + 2*i + 1] = p.y;
      }
    }

    // ---- O += P*V : per key-half, load 4 B-frags then 4 WMMAs ----
    #pragma unroll
    for (int g = 0; g < 2; ++g) {
      #pragma unroll
      for (int np = 0; np < 2; ++np) {
        const int n0 = np * 4;
        v16h b0 = load_half16(vf + (n0 + 0) * 16 * KSTEP + g * 32);
        v16h b1 = load_half16(vf + (n0 + 1) * 16 * KSTEP + g * 32);
        v16h b2 = load_half16(vf + (n0 + 2) * 16 * KSTEP + g * 32);
        v16h b3 = load_half16(vf + (n0 + 3) * 16 * KSTEP + g * 32);
        acc[n0 + 0] = __builtin_amdgcn_wmma_f32_16x16x32_f16(
                          false, pa[g], false, b0, (short)0, acc[n0 + 0], false, false);
        acc[n0 + 1] = __builtin_amdgcn_wmma_f32_16x16x32_f16(
                          false, pa[g], false, b1, (short)0, acc[n0 + 1], false, false);
        acc[n0 + 2] = __builtin_amdgcn_wmma_f32_16x16x32_f16(
                          false, pa[g], false, b2, (short)0, acc[n0 + 2], false, false);
        acc[n0 + 3] = __builtin_amdgcn_wmma_f32_16x16x32_f16(
                          false, pa[g], false, b3, (short)0, acc[n0 + 3], false, false);
      }
    }

    __syncthreads();   // tile s+1 staged; everyone done reading tile s
  }

  // ---- Epilogue: normalize, nontemporal fp32 stores ----
  #pragma unroll
  for (int r = 0; r < 8; ++r) {
    const float inv = 1.0f / lrow[r];
    float* orow = Ob + (size_t)(q0 + r + 8 * hi) * FA_D;
    #pragma unroll
    for (int nt = 0; nt < 8; ++nt)
      __builtin_nontemporal_store(acc[nt][r] * inv, orow + nt * 16 + nB);
  }
}

// ---------------------------------------------------------------------------
extern "C" void kernel_launch(void* const* d_in, const int* in_sizes, int n_in,
                              void* d_out, int out_size, void* d_ws, size_t ws_size,
                              hipStream_t stream) {
  (void)in_sizes; (void)n_in; (void)out_size; (void)d_ws; (void)ws_size;
  const float* q = (const float*)d_in[0];
  const float* k = (const float*)d_in[1];
  const float* v = (const float*)d_in[2];
  float*       o = (float*)d_out;

  dim3 grid(FA_S / QTILE, FA_B * FA_H);   // (16, 32)
  dim3 block(NTHREADS);                   // 8 wave32
  hipLaunchKernelGGL(fa2_fwd_gfx1250, grid, block, 0, stream, q, k, v, o);
}
